// DecompositionModel_33758442947159
// MI455X (gfx1250) — matrix operand order repaired and checked by reading, not compiled, not run on hardware
//
#include <hip/hip_runtime.h>
#include <hip/hip_bf16.h>

typedef __attribute__((ext_vector_type(16))) _Float16 v16h;
typedef __attribute__((ext_vector_type(8)))  _Float16 v8h;
typedef __attribute__((ext_vector_type(8)))  float    v8f;

// ---------------------------------------------------------------------------
// WMMA helpers (CDNA5 wave32, v_wmma_f32_16x16x32_f16)
// ---------------------------------------------------------------------------
__device__ __forceinline__ v8f wmma_f16(v16h a, v16h b, v8f c) {
  return __builtin_amdgcn_wmma_f32_16x16x32_f16(
      /*neg_a=*/false, a, /*neg_b=*/false, b,
      /*c_mod=*/(short)0, c, /*reuse_a=*/false, /*reuse_b=*/false);
}

// A fragment (16x32 f16): lane row = lane&15; halves 0..15 at K=16*hi..16*hi+15
// (hi = lane>>4). Two 16B LDS reads.
__device__ __forceinline__ v16h load_a_frag(const _Float16* tile, int ld, int lane) {
  int row  = lane & 15;
  int koff = (lane >> 4) << 4;
  const _Float16* p = tile + row * ld + koff;
  v8h lo = *(const v8h*)(p);
  v8h hi = *(const v8h*)(p + 8);
  return __builtin_shufflevector(lo, hi, 0,1,2,3,4,5,6,7,8,9,10,11,12,13,14,15);
}

// B fragment (32x16 f16) from tile stored [N][K]: lane col = lane&15,
// halves 0..15 = K = 16*hi .. 16*hi+15 (contiguous 32B).
__device__ __forceinline__ v16h load_b_frag(const _Float16* tile, int ld, int lane) {
  int col  = lane & 15;
  int koff = (lane >> 4) << 4;
  return *(const v16h*)(tile + col * ld + koff);
}

// ---------------------------------------------------------------------------
// CDNA5 async global->LDS DMA (ASYNCcnt path, bypasses VGPRs).
// lds_off: per-lane LDS byte address (low 32 bits of flat shared pointer).
// gptr   : per-lane 16B-aligned global address.
// ---------------------------------------------------------------------------
__device__ __forceinline__ unsigned lds_addr_of(const void* p) {
  return (unsigned)(uintptr_t)p;   // LDS aperture: addr[31:0] is the LDS offset
}
__device__ __forceinline__ void async_copy_b128(unsigned lds_off, const void* gptr) {
  asm volatile("global_load_async_to_lds_b128 %0, %1, off"
               :: "v"(lds_off), "v"(gptr) : "memory");
}
__device__ __forceinline__ void async_wait0() {
  asm volatile("s_wait_asynccnt 0" ::: "memory");
}

// ---------------------------------------------------------------------------
// Small utility kernels
// ---------------------------------------------------------------------------
__global__ void cvt_f16_kernel(const float* __restrict__ in, _Float16* __restrict__ out, int n) {
  int i = blockIdx.x * 256 + threadIdx.x;
  if (i < n) out[i] = (_Float16)in[i];
}

// RoPE features: feats[pos*64 + j], pos=(y*32+x), window 32x32, F=16, THETA=10
__global__ void rope_kernel(float* __restrict__ feats) {
  int idx = blockIdx.x * 256 + threadIdx.x;
  if (idx >= 1024 * 64) return;
  int pos = idx >> 6, j = idx & 63;
  int f = j & 15;
  float inv = powf(10.0f, -(float)f * (1.0f / 16.0f));
  float coord = (j & 32) ? (float)(pos >> 5) : (float)(pos & 31);
  float arg = coord * inv;
  feats[idx] = (j & 16) ? sinf(arg) : cosf(arg);
}

// ---------------------------------------------------------------------------
// Bilinear sample of the 2x-upscaled 512x512 image (half-pixel centers)
// ---------------------------------------------------------------------------
__device__ __forceinline__ float bilin2x(const float* __restrict__ x, int b, int g, int Y, int X) {
  float sy = (Y + 0.5f) * 0.5f - 0.5f;
  float sx = (X + 0.5f) * 0.5f - 0.5f;
  float fy = floorf(sy), fx = floorf(sx);
  int y0 = (int)fy, x0 = (int)fx;
  float wy = sy - fy, wx = sx - fx;
  int y0c = max(y0, 0), y1c = min(y0 + 1, 511);
  int x0c = max(x0, 0), x1c = min(x0 + 1, 511);
  const float* p = x + (((size_t)b * 3 + g) * 512) * 512;
  float v00 = p[y0c * 512 + x0c], v01 = p[y0c * 512 + x1c];
  float v10 = p[y1c * 512 + x0c], v11 = p[y1c * 512 + x1c];
  return (1.0f - wy) * ((1.0f - wx) * v00 + wx * v01) +
         wy * ((1.0f - wx) * v10 + wx * v11);
}

// ---------------------------------------------------------------------------
// K1: bilinear resize + 16x16 grouped conv (stride16) + pos_embed + LayerNorm
// ---------------------------------------------------------------------------
__global__ __launch_bounds__(256) void resize_patch_ln_kernel(
    const float* __restrict__ x, const float* __restrict__ w,
    const float* __restrict__ bias, const float* __restrict__ pos,
    const float* __restrict__ lng, const float* __restrict__ lnb,
    _Float16* __restrict__ out) {
  __shared__ float patch[768];   // 3 groups x 16x16
  __shared__ float red[256];
  int idx = blockIdx.x;
  int b = idx >> 12, yo = (idx >> 6) & 63, xo = idx & 63;
  int t = threadIdx.x;
  for (int k = 0; k < 3; k++) {
    int p = t + 256 * k;
    int g = p >> 8, rem = p & 255, i = rem >> 4, j = rem & 15;
    patch[p] = bilin2x(x, b, g, yo * 16 + i, xo * 16 + j);
  }
  __syncthreads();
  float vals[3], s1 = 0.0f, s2 = 0.0f;
  for (int k = 0; k < 3; k++) {
    int c = t + 256 * k;                         // group = k
    const float* wp = w + (size_t)c * 256;
    const float* pp = patch + k * 256;
    float acc = bias[c];
#pragma unroll 8
    for (int p = 0; p < 256; p++) acc += pp[p] * wp[p];
    acc += pos[((size_t)c * 64 + yo) * 64 + xo];
    vals[k] = acc; s1 += acc; s2 += acc * acc;
  }
  red[t] = s1; __syncthreads();
  for (int off = 128; off; off >>= 1) { if (t < off) red[t] += red[t + off]; __syncthreads(); }
  float mean = red[0] * (1.0f / 768.0f); __syncthreads();
  red[t] = s2; __syncthreads();
  for (int off = 128; off; off >>= 1) { if (t < off) red[t] += red[t + off]; __syncthreads(); }
  float var = red[0] * (1.0f / 768.0f) - mean * mean;
  float rstd = rsqrtf(var + 1e-5f);
  int bw = b * 4 + (yo >> 5) * 2 + (xo >> 5);
  int pn = (yo & 31) * 32 + (xo & 31);
  size_t row = (size_t)bw * 1024 + pn;
  for (int k = 0; k < 3; k++) {
    int c = t + 256 * k;
    float v = (vals[k] - mean) * rstd * lng[c] + lnb[c];
    out[row * 768 + c] = (_Float16)v;
  }
}

// ---------------------------------------------------------------------------
// K2: 8x8 grouped conv (stride8) + LayerNorm -> f16 rows (window order)
// ---------------------------------------------------------------------------
__global__ __launch_bounds__(256) void origin_patch_ln_kernel(
    const float* __restrict__ x, const float* __restrict__ w,
    const float* __restrict__ bias,
    const float* __restrict__ lng, const float* __restrict__ lnb,
    _Float16* __restrict__ out) {
  __shared__ float patch[192];   // 3 groups x 8x8
  __shared__ float red[256];
  int idx = blockIdx.x;
  int b = idx >> 12, yo = (idx >> 6) & 63, xo = idx & 63;
  int t = threadIdx.x;
  if (t < 192) {
    int g = t >> 6, rem = t & 63, i = rem >> 3, j = rem & 7;
    patch[t] = x[(((size_t)b * 3 + g) * 512 + yo * 8 + i) * 512 + xo * 8 + j];
  }
  __syncthreads();
  float vals[3], s1 = 0.0f, s2 = 0.0f;
  for (int k = 0; k < 3; k++) {
    int c = t + 256 * k;
    const float* wp = w + (size_t)c * 64;
    const float* pp = patch + k * 64;
    float acc = bias[c];
#pragma unroll 8
    for (int p = 0; p < 64; p++) acc += pp[p] * wp[p];
    vals[k] = acc; s1 += acc; s2 += acc * acc;
  }
  red[t] = s1; __syncthreads();
  for (int off = 128; off; off >>= 1) { if (t < off) red[t] += red[t + off]; __syncthreads(); }
  float mean = red[0] * (1.0f / 768.0f); __syncthreads();
  red[t] = s2; __syncthreads();
  for (int off = 128; off; off >>= 1) { if (t < off) red[t] += red[t + off]; __syncthreads(); }
  float var = red[0] * (1.0f / 768.0f) - mean * mean;
  float rstd = rsqrtf(var + 1e-5f);
  int bw = b * 4 + (yo >> 5) * 2 + (xo >> 5);
  int pn = (yo & 31) * 32 + (xo & 31);
  size_t row = (size_t)bw * 1024 + pn;
  for (int k = 0; k < 3; k++) {
    int c = t + 256 * k;
    float v = (vals[k] - mean) * rstd * lng[c] + lnb[c];
    out[row * 768 + c] = (_Float16)v;
  }
}

// ---------------------------------------------------------------------------
// WMMA GEMM: C[8192,768] = A[8192,768](f16) @ W[768,768]^T (f16) + bias
// Block tile 128(M)x64(N), 8 waves in 4x2, each wave 32x32 via 2x2 wmma.
// Tiles staged with async global->LDS DMA (ASYNCcnt).
// MODE 0: row-major fp32. MODE 1: scatter to (B=2,768,64,64) NCHW fp32.
// ---------------------------------------------------------------------------
template <int MODE>
__global__ __launch_bounds__(256) void wmma_gemm768_kernel(
    const _Float16* __restrict__ A, const _Float16* __restrict__ W,
    const float* __restrict__ bias, float* __restrict__ C) {
  __shared__ alignas(16) _Float16 sA[128 * 32];
  __shared__ alignas(16) _Float16 sW[64 * 32];
  int t = threadIdx.x;
  int m0 = blockIdx.y * 128;
  int n0 = blockIdx.x * 64;
  int wid = t >> 5, lane = t & 31;
  int wm = wid >> 1, wn = wid & 1;
  v8f acc[2][2] = {};
  for (int kt = 0; kt < 24; kt++) {
    int k0 = kt * 32;
    for (int q = t; q < 512; q += 256) {       // A: 128 rows x 4 x 16B
      int row = q >> 2, part = q & 3;
      async_copy_b128(lds_addr_of(sA + row * 32 + part * 8),
                      A + (size_t)(m0 + row) * 768 + k0 + part * 8);
    }
    {                                          // W: 64 rows x 4 x 16B
      int row = t >> 2, part = t & 3;
      async_copy_b128(lds_addr_of(sW + row * 32 + part * 8),
                      W + (size_t)(n0 + row) * 768 + k0 + part * 8);
    }
    async_wait0();
    __syncthreads();
    v16h a0 = load_a_frag(sA + (wm * 32) * 32, 32, lane);
    v16h a1 = load_a_frag(sA + (wm * 32 + 16) * 32, 32, lane);
    v16h b0 = load_b_frag(sW + (wn * 32) * 32, 32, lane);
    v16h b1 = load_b_frag(sW + (wn * 32 + 16) * 32, 32, lane);
    acc[0][0] = wmma_f16(a0, b0, acc[0][0]);
    acc[0][1] = wmma_f16(a0, b1, acc[0][1]);
    acc[1][0] = wmma_f16(a1, b0, acc[1][0]);
    acc[1][1] = wmma_f16(a1, b1, acc[1][1]);
    __syncthreads();
  }
  int hi = lane >> 4, nl = lane & 15;
#pragma unroll
  for (int i = 0; i < 2; i++)
#pragma unroll
    for (int j = 0; j < 2; j++) {
      int n = n0 + wn * 32 + 16 * j + nl;
      float bv = bias[n];
#pragma unroll
      for (int r = 0; r < 8; r++) {
        int m = m0 + wm * 32 + 16 * i + r + 8 * hi;
        float v = acc[i][j][r] + bv;
        if (MODE == 0) {
          C[(size_t)m * 768 + n] = v;
        } else {
          int b = m >> 12, y = (m >> 6) & 63, xx = m & 63;
          C[(((size_t)b * 768 + n) * 64 + y) * 64 + xx] = v;
        }
      }
    }
}

// ---------------------------------------------------------------------------
// K5/K6: per (bw,pos,head): L2-normalize [96 proj | 64 rope] -> f16 (K=160);
// optionally also emit raw head values (tmp_b) as f16 [bh][1024][96].
// ---------------------------------------------------------------------------
__global__ __launch_bounds__(256) void finalize_kernel(
    const float* __restrict__ G, const float* __restrict__ feats,
    _Float16* __restrict__ Qn, _Float16* __restrict__ Vt) {
  int wid = threadIdx.x >> 5, lane = threadIdx.x & 31;
  int task = blockIdx.x * 8 + wid;           // 65536 = 8*1024*8
  int h = task & 7, pos = (task >> 3) & 1023, bw = task >> 13;
  size_t grow = ((size_t)bw * 1024 + pos) * 768 + h * 96;
  float v0 = G[grow + lane], v1 = G[grow + lane + 32], v2 = G[grow + lane + 64];
  float f0 = feats[pos * 64 + lane], f1 = feats[pos * 64 + lane + 32];
  float s = v0 * v0 + v1 * v1 + v2 * v2 + f0 * f0 + f1 * f1;
  for (int off = 16; off; off >>= 1) s += __shfl_xor(s, off, 32);
  float inv = 1.0f / fmaxf(sqrtf(s), 1e-12f);
  size_t bh = (size_t)bw * 8 + h;
  size_t orow = (bh * 1024 + pos) * 160;
  Qn[orow + lane]      = (_Float16)(v0 * inv);
  Qn[orow + lane + 32] = (_Float16)(v1 * inv);
  Qn[orow + lane + 64] = (_Float16)(v2 * inv);
  Qn[orow + 96 + lane]      = (_Float16)(f0 * inv);
  Qn[orow + 96 + lane + 32] = (_Float16)(f1 * inv);
  if (Vt) {
    size_t vrow = (bh * 1024 + pos) * 96;
    Vt[vrow + lane]      = (_Float16)v0;
    Vt[vrow + lane + 32] = (_Float16)v1;
    Vt[vrow + lane + 64] = (_Float16)v2;
  }
}

// ---------------------------------------------------------------------------
// K7: fused attention. Per (bh, 64-row q tile): loop 64-row b chunks:
//   S = Qn(64x160) x Bn_chunk^T  -> write attn fp32; f16 S via LDS;
//   accO += S(64x64) x V(64x96)  (V transposed in LDS).
// Q/B tiles staged with async global->LDS DMA.
// ---------------------------------------------------------------------------
__global__ __launch_bounds__(256) void attn_kernel(
    const _Float16* __restrict__ Q, const _Float16* __restrict__ Bn,
    const _Float16* __restrict__ V, float* __restrict__ attn_out,
    _Float16* __restrict__ outA) {
  __shared__ alignas(16) _Float16 sQ[64 * 160];   // 20KB
  __shared__ alignas(16) _Float16 sB[64 * 160];   // 20KB
  __shared__ alignas(16) _Float16 sVt[96 * 64];   // 12KB (transposed: [d][k])
  __shared__ alignas(16) _Float16 sS[64 * 64];    // 8KB
  int bh = blockIdx.y;
  int qm0 = blockIdx.x * 64;
  int t = threadIdx.x, wid = t >> 5, lane = t & 31;
  int hi = lane >> 4, nl = lane & 15;
  int wmS = wid >> 2, wnS = wid & 3;   // S tile: 2(M:32) x 4(N:16)
  int wmO = wid >> 1, wnO = wid & 1;   // out tile: 4(M:16) x 2(N:48)
  for (int q = t; q < 64 * 20; q += 256) {        // async stage sQ (once)
    int row = q / 20, part = q % 20;
    async_copy_b128(lds_addr_of(sQ + row * 160 + part * 8),
                    Q + ((size_t)bh * 1024 + qm0 + row) * 160 + part * 8);
  }
  v8f accO[3] = {};
  for (int kb = 0; kb < 16; kb++) {
    int kr0 = kb * 64;
    __syncthreads();                              // protect sB/sVt/sS reuse
    for (int q = t; q < 64 * 20; q += 256) {      // async stage sB chunk
      int row = q / 20, part = q % 20;
      async_copy_b128(lds_addr_of(sB + row * 160 + part * 8),
                      Bn + ((size_t)bh * 1024 + kr0 + row) * 160 + part * 8);
    }
    for (int q = t; q < 64 * 48; q += 256) {      // transpose V chunk (4B loads)
      int kr = q / 48, dp = q - kr * 48;
      union { unsigned u; _Float16 h[2]; } cv;
      cv.u = *(const unsigned*)(V + ((size_t)bh * 1024 + kr0 + kr) * 96 + dp * 2);
      sVt[(dp * 2) * 64 + kr]     = cv.h[0];
      sVt[(dp * 2 + 1) * 64 + kr] = cv.h[1];
    }
    async_wait0();                                // sQ (first iter) + sB landed
    __syncthreads();
    v8f accS[2] = {};
#pragma unroll
    for (int kt = 0; kt < 5; kt++) {
      v16h b0 = load_b_frag(sB + (wnS * 16) * 160 + kt * 32, 160, lane);
      v16h a0 = load_a_frag(sQ + (wmS * 32) * 160 + kt * 32, 160, lane);
      v16h a1 = load_a_frag(sQ + (wmS * 32 + 16) * 160 + kt * 32, 160, lane);
      accS[0] = wmma_f16(a0, b0, accS[0]);
      accS[1] = wmma_f16(a1, b0, accS[1]);
    }
#pragma unroll
    for (int i = 0; i < 2; i++)
#pragma unroll
      for (int r = 0; r < 8; r++) {
        int ml = wmS * 32 + 16 * i + r + 8 * hi;
        int nc = wnS * 16 + nl;
        float v = accS[i][r];
        attn_out[((size_t)bh * 1024 + qm0 + ml) * 1024 + kr0 + nc] = v;
        sS[ml * 64 + nc] = (_Float16)v;
      }
    __syncthreads();
#pragma unroll
    for (int kt = 0; kt < 2; kt++) {
      v16h a = load_a_frag(sS + (wmO * 16) * 64 + kt * 32, 64, lane);
#pragma unroll
      for (int j = 0; j < 3; j++) {
        v16h bf = load_b_frag(sVt + (wnO * 48 + 16 * j) * 64 + kt * 32, 64, lane);
        accO[j] = wmma_f16(a, bf, accO[j]);
      }
    }
  }
  // Epilogue: out_heads[bh][pos][d] -> row (b, y, x), col h*96+d (f16)
  int bw = bh >> 3, h = bh & 7;
  int bb = bw >> 2, wy = (bw >> 1) & 1, wx = bw & 1;
#pragma unroll
  for (int j = 0; j < 3; j++)
#pragma unroll
    for (int r = 0; r < 8; r++) {
      int pos = qm0 + wmO * 16 + r + 8 * hi;
      int y = wy * 32 + (pos >> 5);
      int xx = wx * 32 + (pos & 31);
      size_t R = (size_t)bb * 4096 + y * 64 + xx;
      int col = h * 96 + wnO * 48 + 16 * j + nl;
      outA[R * 768 + col] = (_Float16)accO[j][r];
    }
}

// ---------------------------------------------------------------------------
// Host launcher
// ---------------------------------------------------------------------------
extern "C" void kernel_launch(void* const* d_in, const int* in_sizes, int n_in,
                              void* d_out, int out_size, void* d_ws, size_t ws_size,
                              hipStream_t stream) {
  (void)in_sizes; (void)n_in; (void)out_size; (void)ws_size;
  const float* x        = (const float*)d_in[0];
  const float* origin_w = (const float*)d_in[1];
  const float* origin_b = (const float*)d_in[2];
  const float* resize_w = (const float*)d_in[3];
  const float* resize_b = (const float*)d_in[4];
  const float* pos_emb  = (const float*)d_in[5];
  const float* nb_g = (const float*)d_in[6];
  const float* nb_b = (const float*)d_in[7];
  const float* nq_g = (const float*)d_in[8];
  const float* nq_b = (const float*)d_in[9];
  const float* qw = (const float*)d_in[10];
  const float* qb = (const float*)d_in[11];
  const float* bw = (const float*)d_in[12];
  const float* bb = (const float*)d_in[13];
  const float* pw = (const float*)d_in[14];
  const float* pb = (const float*)d_in[15];

  float* out_f32  = (float*)d_out;                  // (2,768,64,64)
  float* attn_f32 = out_f32 + 2ull * 768 * 64 * 64; // (8,8,1024,1024)

  char* ws = (char*)d_ws;
  size_t off = 0;
  auto take = [&](size_t bytes) { size_t r = off; off += (bytes + 255) & ~255ull; return r; };
  const size_t W16   = 768ull * 768 * 2;
  _Float16* qw16   = (_Float16*)(ws + take(W16));
  _Float16* bw16   = (_Float16*)(ws + take(W16));
  _Float16* pw16   = (_Float16*)(ws + take(W16));
  float*    rope   = (float*)   (ws + take(1024ull * 64 * 4));
  _Float16* rp_ln  = (_Float16*)(ws + take(8192ull * 768 * 2));  // reused as outA
  _Float16* op_ln  = (_Float16*)(ws + take(8192ull * 768 * 2));
  float*    gemm32 = (float*)   (ws + take(8192ull * 768 * 4));  // shared q/b fp32
  _Float16* qn     = (_Float16*)(ws + take(64ull * 1024 * 160 * 2));
  _Float16* bn     = (_Float16*)(ws + take(64ull * 1024 * 160 * 2));
  _Float16* vt     = (_Float16*)(ws + take(64ull * 1024 * 96 * 2));
  _Float16* outA   = rp_ln;  // rp_ln is dead before attn_kernel runs

  const int NW = 768 * 768;
  cvt_f16_kernel<<<(NW + 255) / 256, 256, 0, stream>>>(qw, qw16, NW);
  cvt_f16_kernel<<<(NW + 255) / 256, 256, 0, stream>>>(bw, bw16, NW);
  cvt_f16_kernel<<<(NW + 255) / 256, 256, 0, stream>>>(pw, pw16, NW);
  rope_kernel<<<256, 256, 0, stream>>>(rope);

  resize_patch_ln_kernel<<<8192, 256, 0, stream>>>(x, resize_w, resize_b, pos_emb,
                                                   nq_g, nq_b, rp_ln);
  origin_patch_ln_kernel<<<8192, 256, 0, stream>>>(x, origin_w, origin_b,
                                                   nb_g, nb_b, op_ln);

  dim3 ggrid(12, 64);  // 768/64 N-blocks, 8192/128 M-blocks
  wmma_gemm768_kernel<0><<<ggrid, 256, 0, stream>>>(rp_ln, qw16, qb, gemm32);
  finalize_kernel<<<8192, 256, 0, stream>>>(gemm32, rope, qn, (_Float16*)nullptr);
  wmma_gemm768_kernel<0><<<ggrid, 256, 0, stream>>>(op_ln, bw16, bb, gemm32);
  finalize_kernel<<<8192, 256, 0, stream>>>(gemm32, rope, bn, vt);

  dim3 agrid(16, 64);  // 1024/64 q-tiles x 64 (bw*heads)
  attn_kernel<<<agrid, 256, 0, stream>>>(qn, bn, vt, attn_f32, outA);

  wmma_gemm768_kernel<1><<<ggrid, 256, 0, stream>>>(outA, pw16, pb, out_f32);
}
